// MaskedMHCA_49263274885551
// MI455X (gfx1250) — matrix operand order repaired and checked by reading, not compiled, not run on hardware
//
#include <hip/hip_runtime.h>
#include <hip/hip_bf16.h>

typedef __attribute__((ext_vector_type(16))) _Float16 v16h;
typedef __attribute__((ext_vector_type(8)))  _Float16 v8h;
typedef __attribute__((ext_vector_type(4)))  _Float16 v4h;
typedef __attribute__((ext_vector_type(8)))  float    v8f;

#define B_   2
#define C_   1024
#define T_   2048
#define H_   16
#define D_   64
#define NE_  (B_*C_*T_)          // 4194304
#define SCALE_ 0.125f            // 1/sqrt(64)
#define EPS_   1e-5f
#define NEG_BIG (-1.0e30f)

// ---------------------------------------------------------------------------
// WMMA helpers (CDNA5 16x16x32 f16 -> f32, wave32 fragments, contiguous k = 16*hi + j per lane half)
// ---------------------------------------------------------------------------
__device__ __forceinline__ v8f wmma16(v16h a, v16h b, v8f c) {
    return __builtin_amdgcn_wmma_f32_16x16x32_f16(
        /*neg_a=*/false, a, /*neg_b=*/false, b,
        /*c_mod=*/(short)0, c, /*reuse_a=*/false, /*reuse_b=*/false);
}

__device__ __forceinline__ v8f zero8() {
    v8f z = {0.f,0.f,0.f,0.f,0.f,0.f,0.f,0.f};
    return z;
}

// A fragment (16 M x 32 K): per-lane halves are 16 contiguous k (k = 16*hi + j)
__device__ __forceinline__ v16h load_a_frag(const _Float16* base, int ld, int m0, int k0) {
    int lane = threadIdx.x & 31;
    int m  = m0 + (lane & 15);
    int hi = (lane >> 4) & 1;
    const _Float16* p = base + m*ld + k0 + hi*16;
    v16h a;
#pragma unroll
    for (int j = 0; j < 8; ++j) {
        a[j]     = p[j];
        a[j + 8] = p[8 + j];
    }
    return a;
}

// B fragment (32 K x 16 N) gathered from an [n][k] (transposed) LDS tile:
// 16 contiguous halves per lane = 2x ds_load_b128
__device__ __forceinline__ v16h load_b_frag_nk(const _Float16* base, int ld, int n0, int k0) {
    int lane = threadIdx.x & 31;
    int n  = n0 + (lane & 15);
    int hi = (lane >> 4) & 1;
    const _Float16* p = base + n*ld + k0 + hi*16;
    v16h b;
#pragma unroll
    for (int j = 0; j < 16; ++j) b[j] = p[j];
    return b;
}

// ---------------------------------------------------------------------------
// K0: f32 -> f16 weight conversion
// ---------------------------------------------------------------------------
__global__ void wcvt_kernel(const float* __restrict__ src, _Float16* __restrict__ dst, int n) {
    int i = blockIdx.x * blockDim.x + threadIdx.x;
    if (i < n) dst[i] = (_Float16)src[i];
}

// ---------------------------------------------------------------------------
// K1: fused depthwise conv3 (pad 1) * mask + temporal LayerNorm over C, x3
//     wave-level shfl_xor reductions + one LDS partial stage (1 barrier)
// ---------------------------------------------------------------------------
__global__ __launch_bounds__(256) void conv_ln_kernel(
    const float* __restrict__ x, const unsigned char* __restrict__ mask,
    const float* __restrict__ wq, const float* __restrict__ wk, const float* __restrict__ wv,
    const float* __restrict__ gq, const float* __restrict__ bq,
    const float* __restrict__ gk, const float* __restrict__ bk,
    const float* __restrict__ gv, const float* __restrict__ bv,
    _Float16* __restrict__ xq, _Float16* __restrict__ xk, _Float16* __restrict__ xv)
{
    int bt = blockIdx.x;
    int b = bt / T_, t = bt % T_;
    int tid = threadIdx.x;
    float mf = mask[b*T_ + t] ? 1.f : 0.f;

    float yq[4], yk[4], yv[4];
    float vals[6] = {0.f,0.f,0.f,0.f,0.f,0.f};   // sq,sqq,sk,skk,sv,svv
#pragma unroll
    for (int i = 0; i < 4; ++i) {
        int c = tid + i*256;
        const float* xr = x + ((size_t)b*C_ + c)*T_;
        float xm = (t > 0)     ? xr[t-1] : 0.f;
        float x0 = xr[t];
        float xp = (t < T_-1)  ? xr[t+1] : 0.f;
        float q = (wq[c*3]*xm + wq[c*3+1]*x0 + wq[c*3+2]*xp) * mf;
        float k = (wk[c*3]*xm + wk[c*3+1]*x0 + wk[c*3+2]*xp) * mf;
        float v = (wv[c*3]*xm + wv[c*3+1]*x0 + wv[c*3+2]*xp) * mf;
        yq[i]=q; yk[i]=k; yv[i]=v;
        vals[0]+=q; vals[1]+=q*q; vals[2]+=k; vals[3]+=k*k; vals[4]+=v; vals[5]+=v*v;
    }

    // wave32 butterfly reduction (no LDS, no barriers)
#pragma unroll
    for (int off = 16; off > 0; off >>= 1) {
#pragma unroll
        for (int j = 0; j < 6; ++j) vals[j] += __shfl_xor(vals[j], off);
    }
    __shared__ float sPart[8][6];
    int wid = tid >> 5;
    if ((tid & 31) == 0) {
#pragma unroll
        for (int j = 0; j < 6; ++j) sPart[wid][j] = vals[j];
    }
    __syncthreads();
    float tot[6];
#pragma unroll
    for (int j = 0; j < 6; ++j) {
        float s = 0.f;
#pragma unroll
        for (int w = 0; w < 8; ++w) s += sPart[w][j];
        tot[j] = s;
    }

    const float invC = 1.f / (float)C_;
    float mq = tot[0]*invC, vq  = tot[1]*invC - mq*mq;
    float mk = tot[2]*invC, vk  = tot[3]*invC - mk*mk;
    float mv = tot[4]*invC, vv2 = tot[5]*invC - mv*mv;
    float rq = rsqrtf(vq  + EPS_);
    float rk = rsqrtf(vk  + EPS_);
    float rv = rsqrtf(vv2 + EPS_);

#pragma unroll
    for (int i = 0; i < 4; ++i) {
        int c = tid + i*256;
        size_t o = ((size_t)b*C_ + c)*T_ + t;
        xq[o] = (_Float16)((yq[i]-mq)*rq*gq[c] + bq[c]);
        xk[o] = (_Float16)((yk[i]-mk)*rk*gk[c] + bk[c]);
        xv[o] = (_Float16)((yv[i]-mv)*rv*gv[c] + bv[c]);
    }
}

// ---------------------------------------------------------------------------
// K2: tiled WMMA GEMM  Y[o,n] = sum_c W[o,c] * X[c,n] + bias[o]
//   128(M) x 128(N) block tile, double-buffered LDS, 128-bit staging
//   (all 4 global loads issued before any LDS store -> loads in flight).
//   mode 0: out f16 [B,H,T,D]   mode 1: out f32 [B,C,T] * mask
// ---------------------------------------------------------------------------
__global__ __launch_bounds__(256) void gemm_wmma_kernel(
    const _Float16* __restrict__ W, const _Float16* __restrict__ X,
    const float* __restrict__ bias,
    _Float16* __restrict__ out16, float* __restrict__ out32,
    const unsigned char* __restrict__ mask, int mode)
{
    __shared__ __align__(16) _Float16 sA [2][128][40];   // [m][k]
    __shared__ __align__(16) _Float16 sBt[2][128][40];   // [n][k] (transposed)

    int m0  = blockIdx.x * 128;
    int n0  = blockIdx.y * 128;
    int tid = threadIdx.x;
    int wid = tid >> 5;

    int bb0 = n0 >> 11;            // 128-wide N tile lives in one batch (T%128==0)
    int tt0 = n0 & (T_-1);
    const _Float16* Xb = X + (size_t)bb0*C_*T_ + tt0;

    int mA  = tid >> 2,  c8A = (tid & 3)  * 8;   // A: 128x32, 2 chunks/thread
    int kB  = tid >> 4,  c8B = (tid & 15) * 8;   // B: 32x128, 2 chunks/thread

    auto stage = [&](int buf, int kb) {
        // phase 1: issue all 4 global_load_b128 (kept in flight together)
        v8h a0 = *(const v8h*)&W[(size_t)(m0 + mA     )*C_ + kb + c8A];
        v8h a1 = *(const v8h*)&W[(size_t)(m0 + mA + 64)*C_ + kb + c8A];
        v8h b0 = *(const v8h*)&Xb[(size_t)(kb + kB     )*T_ + c8B];
        v8h b1 = *(const v8h*)&Xb[(size_t)(kb + kB + 16)*T_ + c8B];
        // phase 2: LDS stores
        *(v8h*)&sA[buf][mA     ][c8A] = a0;
        *(v8h*)&sA[buf][mA + 64][c8A] = a1;
#pragma unroll
        for (int j = 0; j < 8; ++j) sBt[buf][c8B + j][kB     ] = b0[j];
#pragma unroll
        for (int j = 0; j < 8; ++j) sBt[buf][c8B + j][kB + 16] = b1[j];
    };

    v8f acc[8];
#pragma unroll
    for (int i = 0; i < 8; ++i) acc[i] = zero8();

    stage(0, 0);
    int cur = 0;
    for (int kb = 0; kb < C_; kb += 32) {
        __syncthreads();
        if (kb + 32 < C_) stage(cur ^ 1, kb + 32);   // overlap with compute
        v16h a = load_a_frag(&sA[cur][0][0], 40, wid*16, 0);
#pragma unroll
        for (int g = 0; g < 2; ++g) {
            v16h bf0 = load_b_frag_nk(&sBt[cur][0][0], 40, (g*4+0)*16, 0);
            v16h bf1 = load_b_frag_nk(&sBt[cur][0][0], 40, (g*4+1)*16, 0);
            v16h bf2 = load_b_frag_nk(&sBt[cur][0][0], 40, (g*4+2)*16, 0);
            v16h bf3 = load_b_frag_nk(&sBt[cur][0][0], 40, (g*4+3)*16, 0);
            acc[g*4+0] = wmma16(a, bf0, acc[g*4+0]);
            acc[g*4+1] = wmma16(a, bf1, acc[g*4+1]);
            acc[g*4+2] = wmma16(a, bf2, acc[g*4+2]);
            acc[g*4+3] = wmma16(a, bf3, acc[g*4+3]);
        }
        cur ^= 1;
    }

    int lane = tid & 31;
    int hi = (lane >> 4) & 1;
    int nn = lane & 15;
    int mb = m0 + wid*16 + hi*8;                   // 8 consecutive m per lane
    float4 blo = *(const float4*)&bias[mb];
    float4 bhi = *(const float4*)&bias[mb + 4];
    float bv8[8] = {blo.x,blo.y,blo.z,blo.w,bhi.x,bhi.y,bhi.z,bhi.w};

    if (mode == 0) {
        int h = mb >> 6, d0 = mb & 63;             // consecutive m -> consecutive d
#pragma unroll
        for (int nt = 0; nt < 8; ++nt) {
            int n = n0 + nt*16 + nn;
            int bb = n >> 11, tt = n & (T_-1);
            v8h pk;
#pragma unroll
            for (int r = 0; r < 8; ++r) pk[r] = (_Float16)(acc[nt][r] + bv8[r]);
            *(v8h*)&out16[(((size_t)bb*H_ + h)*T_ + tt)*D_ + d0] = pk;
        }
    } else {
#pragma unroll
        for (int nt = 0; nt < 8; ++nt) {
            int n = n0 + nt*16 + nn;
            int bb = n >> 11, tt = n & (T_-1);
            float mf = mask[bb*T_ + tt] ? 1.f : 0.f;
#pragma unroll
            for (int r = 0; r < 8; ++r) {
                // lanes 0..15 write 16 consecutive floats -> coalesced lines
                out32[((size_t)bb*C_ + mb + r)*T_ + tt] = (acc[nt][r] + bv8[r]) * mf;
            }
        }
    }
}

// ---------------------------------------------------------------------------
// K3: flash attention per (b, h, 64-query tile), 256 threads = 8 waves
//   Q,K,V: [B,H,T,D] f16;  output attn: [B,C,T] f16 (c = h*64 + d)
// ---------------------------------------------------------------------------
__global__ __launch_bounds__(256) void attn_kernel(
    const _Float16* __restrict__ Q, const _Float16* __restrict__ K,
    const _Float16* __restrict__ V, const unsigned char* __restrict__ mask,
    _Float16* __restrict__ O)
{
    __shared__ __align__(16) _Float16 sQ [64][72];   // [tq][d]
    __shared__ __align__(16) _Float16 sK [64][72];   // [s][d]
    __shared__ __align__(16) _Float16 sVt[64][72];   // [d][s] (transposed)
    __shared__ __align__(16) float    sS [64][68];
    __shared__ __align__(16) _Float16 sP [64][72];   // [tq][s]
    __shared__ float sM[64], sL[64], sAl[64];

    int bx = blockIdx.x;
    int qt = bx & 31;            // T/64 = 32 query tiles
    int h  = (bx >> 5) & (H_-1);
    int b  = bx >> 9;
    int tid  = threadIdx.x;
    int wid  = tid >> 5;
    int lane = tid & 31;
    int hi   = (lane >> 4) & 1;
    int nn   = lane & 15;

    const _Float16* Qb = Q + ((size_t)b*H_ + h)*T_*D_;
    const _Float16* Kb = K + ((size_t)b*H_ + h)*T_*D_;
    const _Float16* Vb = V + ((size_t)b*H_ + h)*T_*D_;

    int rS  = tid >> 3, c8S = (tid & 7) * 8;     // tile staging: 2 chunks/thread

    // stage Q tile (loads first, then stores)
    {
        v8h q0 = *(const v8h*)&Qb[(size_t)(qt*64 + rS     )*D_ + c8S];
        v8h q1 = *(const v8h*)&Qb[(size_t)(qt*64 + rS + 32)*D_ + c8S];
        *(v8h*)&sQ[rS     ][c8S] = q0;
        *(v8h*)&sQ[rS + 32][c8S] = q1;
    }
    if (tid < 64) { sM[tid] = NEG_BIG; sL[tid] = 0.f; }

    // two 16x16 tiles per wave over the 64x64 tile grid
    int id0 = wid*2, id1 = wid*2 + 1;
    int mt0 = id0 >> 2, nt0 = id0 & 3;
    int mt1 = id1 >> 2, nt1 = id1 & 3;
    v8f o0 = zero8(), o1 = zero8();

    for (int st = 0; st < T_/64; ++st) {
        __syncthreads();   // previous iteration finished with sK/sVt/sP
        {
            // issue all 4 global_load_b128 before any LDS store
            v8h k0 = *(const v8h*)&Kb[(size_t)(st*64 + rS     )*D_ + c8S];
            v8h k1 = *(const v8h*)&Kb[(size_t)(st*64 + rS + 32)*D_ + c8S];
            v8h v0 = *(const v8h*)&Vb[(size_t)(st*64 + rS     )*D_ + c8S];
            v8h v1 = *(const v8h*)&Vb[(size_t)(st*64 + rS + 32)*D_ + c8S];
            *(v8h*)&sK[rS     ][c8S] = k0;
            *(v8h*)&sK[rS + 32][c8S] = k1;
#pragma unroll
            for (int j = 0; j < 8; ++j) sVt[c8S + j][rS     ] = v0[j];
#pragma unroll
            for (int j = 0; j < 8; ++j) sVt[c8S + j][rS + 32] = v1[j];
        }
        if (st + 1 < T_/64) {                // prefetch next tile's rows
            if (tid < 64)       __builtin_prefetch(&Kb[(size_t)((st+1)*64 + tid)*D_], 0, 0);
            else if (tid < 128) __builtin_prefetch(&Vb[(size_t)((st+1)*64 + (tid-64))*D_], 0, 0);
        }
        __syncthreads();

        // S = (Q * K^T) * SCALE, masked ; B-frags from sK[s][d] ([n][k] layout)
        v8f s0 = zero8(), s1 = zero8();
#pragma unroll
        for (int kk = 0; kk < 2; ++kk) {
            v16h a0 = load_a_frag   (&sQ[0][0], 72, mt0*16, kk*32);
            v16h b0 = load_b_frag_nk(&sK[0][0], 72, nt0*16, kk*32);
            v16h a1 = load_a_frag   (&sQ[0][0], 72, mt1*16, kk*32);
            v16h b1 = load_b_frag_nk(&sK[0][0], 72, nt1*16, kk*32);
            s0 = wmma16(a0, b0, s0);
            s1 = wmma16(a1, b1, s1);
        }
        {
            int n0c = nt0*16 + nn;
            int n1c = nt1*16 + nn;
            bool mk0 = mask[b*T_ + st*64 + n0c];
            bool mk1 = mask[b*T_ + st*64 + n1c];
#pragma unroll
            for (int r = 0; r < 8; ++r) {
                sS[mt0*16 + hi*8 + r][n0c] = mk0 ? s0[r]*SCALE_ : NEG_BIG;
                sS[mt1*16 + hi*8 + r][n1c] = mk1 ? s1[r]*SCALE_ : NEG_BIG;
            }
        }
        __syncthreads();

        // online softmax: one thread per query row, vectorized row scans
        if (tid < 64) {
            int row = tid;
            float mold = sM[row];
            float tm = NEG_BIG;
#pragma unroll
            for (int j = 0; j < 16; ++j) {
                float4 v = *(const float4*)&sS[row][4*j];
                tm = fmaxf(tm, fmaxf(fmaxf(v.x, v.y), fmaxf(v.z, v.w)));
            }
            float mnew  = fmaxf(mold, tm);
            float alpha = __expf(mold - mnew);
            float sum = 0.f;
#pragma unroll
            for (int j = 0; j < 16; ++j) {
                float4 v = *(const float4*)&sS[row][4*j];
                float p0 = __expf(v.x - mnew);
                float p1 = __expf(v.y - mnew);
                float p2 = __expf(v.z - mnew);
                float p3 = __expf(v.w - mnew);
                sum += (p0 + p1) + (p2 + p3);
                *(v4h*)&sP[row][4*j] =
                    (v4h){(_Float16)p0, (_Float16)p1, (_Float16)p2, (_Float16)p3};
            }
            sL[row]  = sL[row]*alpha + sum;
            sM[row]  = mnew;
            sAl[row] = alpha;
        }
        __syncthreads();

        // O = O*alpha + P @ V ; B-frags from sVt[d][s] ([n][k] layout)
#pragma unroll
        for (int r = 0; r < 8; ++r) {
            o0[r] *= sAl[mt0*16 + hi*8 + r];
            o1[r] *= sAl[mt1*16 + hi*8 + r];
        }
#pragma unroll
        for (int kk = 0; kk < 2; ++kk) {
            v16h a0 = load_a_frag   (&sP [0][0], 72, mt0*16, kk*32);
            v16h b0 = load_b_frag_nk(&sVt[0][0], 72, nt0*16, kk*32);
            v16h a1 = load_a_frag   (&sP [0][0], 72, mt1*16, kk*32);
            v16h b1 = load_b_frag_nk(&sVt[0][0], 72, nt1*16, kk*32);
            o0 = wmma16(a0, b0, o0);
            o1 = wmma16(a1, b1, o1);
        }
    }
    __syncthreads();

    // finalize: divide by row sums; 8 per-lane values are 8 consecutive t's
    // at fixed d -> single 128-bit store per tile per lane
    {
        int d0 = nt0*16 + nn;
        v8h pk;
#pragma unroll
        for (int r = 0; r < 8; ++r) {
            float l = sL[mt0*16 + hi*8 + r];
            pk[r] = (_Float16)(o0[r] * ((l > 0.f) ? 1.f/l : 0.f));
        }
        *(v8h*)&O[((size_t)b*C_ + h*D_ + d0)*T_ + qt*64 + mt0*16 + hi*8] = pk;
    }
    {
        int d1 = nt1*16 + nn;
        v8h pk;
#pragma unroll
        for (int r = 0; r < 8; ++r) {
            float l = sL[mt1*16 + hi*8 + r];
            pk[r] = (_Float16)(o1[r] * ((l > 0.f) ? 1.f/l : 0.f));
        }
        *(v8h*)&O[((size_t)b*C_ + h*D_ + d1)*T_ + qt*64 + mt1*16 + hi*8] = pk;
    }
}

// ---------------------------------------------------------------------------
// K4: emit qx_mask (second tuple output) as floats after the main output
// ---------------------------------------------------------------------------
__global__ void mask_out_kernel(const unsigned char* __restrict__ mask, float* __restrict__ out) {
    int i = blockIdx.x * blockDim.x + threadIdx.x;
    if (i < B_*T_) out[(size_t)NE_ + i] = mask[i] ? 1.f : 0.f;
}

// ---------------------------------------------------------------------------
extern "C" void kernel_launch(void* const* d_in, const int* in_sizes, int n_in,
                              void* d_out, int out_size, void* d_ws, size_t ws_size,
                              hipStream_t stream)
{
    const float*          x     = (const float*)d_in[0];
    const unsigned char*  mask  = (const unsigned char*)d_in[1];
    const float* wq_dw = (const float*)d_in[2];
    const float* wk_dw = (const float*)d_in[3];
    const float* wv_dw = (const float*)d_in[4];
    const float* ln_q_w = (const float*)d_in[5];  const float* ln_q_b = (const float*)d_in[6];
    const float* ln_k_w = (const float*)d_in[7];  const float* ln_k_b = (const float*)d_in[8];
    const float* ln_v_w = (const float*)d_in[9];  const float* ln_v_b = (const float*)d_in[10];
    const float* Wq = (const float*)d_in[11]; const float* bq = (const float*)d_in[12];
    const float* Wk = (const float*)d_in[13]; const float* bk = (const float*)d_in[14];
    const float* Wv = (const float*)d_in[15]; const float* bv = (const float*)d_in[16];
    const float* Wp = (const float*)d_in[17]; const float* bp = (const float*)d_in[18];

    char* ws = (char*)d_ws;
    const size_t f16buf = (size_t)NE_ * sizeof(_Float16);     // 8 MB
    const size_t wbuf   = (size_t)C_ * C_ * sizeof(_Float16); // 2 MB
    _Float16* xnq  = (_Float16*)(ws + 0*f16buf);
    _Float16* xnk  = (_Float16*)(ws + 1*f16buf);
    _Float16* xnv  = (_Float16*)(ws + 2*f16buf);
    _Float16* qb   = (_Float16*)(ws + 3*f16buf);
    _Float16* kb   = (_Float16*)(ws + 4*f16buf);
    _Float16* vb   = (_Float16*)(ws + 5*f16buf);
    _Float16* ab   = (_Float16*)(ws + 6*f16buf);
    _Float16* Wq16 = (_Float16*)(ws + 7*f16buf + 0*wbuf);
    _Float16* Wk16 = (_Float16*)(ws + 7*f16buf + 1*wbuf);
    _Float16* Wv16 = (_Float16*)(ws + 7*f16buf + 2*wbuf);
    _Float16* Wp16 = (_Float16*)(ws + 7*f16buf + 3*wbuf);

    const int nw = C_*C_;
    dim3 cvt_grid((nw + 255)/256);
    wcvt_kernel<<<cvt_grid, 256, 0, stream>>>(Wq, Wq16, nw);
    wcvt_kernel<<<cvt_grid, 256, 0, stream>>>(Wk, Wk16, nw);
    wcvt_kernel<<<cvt_grid, 256, 0, stream>>>(Wv, Wv16, nw);
    wcvt_kernel<<<cvt_grid, 256, 0, stream>>>(Wp, Wp16, nw);

    conv_ln_kernel<<<B_*T_, 256, 0, stream>>>(
        x, mask, wq_dw, wk_dw, wv_dw,
        ln_q_w, ln_q_b, ln_k_w, ln_k_b, ln_v_w, ln_v_b,
        xnq, xnk, xnv);

    dim3 ggrid(C_/128, (B_*T_)/128);
    gemm_wmma_kernel<<<ggrid, 256, 0, stream>>>(Wq16, xnq, bq, qb, nullptr, nullptr, 0);
    gemm_wmma_kernel<<<ggrid, 256, 0, stream>>>(Wk16, xnk, bk, kb, nullptr, nullptr, 0);
    gemm_wmma_kernel<<<ggrid, 256, 0, stream>>>(Wv16, xnv, bv, vb, nullptr, nullptr, 0);

    attn_kernel<<<B_*H_*(T_/64), 256, 0, stream>>>(qb, kb, vb, mask, ab);

    gemm_wmma_kernel<<<ggrid, 256, 0, stream>>>(Wp16, ab, bp, nullptr, (float*)d_out, mask, 1);

    mask_out_kernel<<<(B_*T_ + 255)/256, 256, 0, stream>>>(mask, (float*)d_out);
}
